// AFC_22162031247480
// MI455X (gfx1250) — matrix operand order repaired and checked by reading, not compiled, not run on hardware
//
#include <hip/hip_runtime.h>

// ---------------------------------------------------------------------------
// Types for CDNA5 WMMA (wave32)
// ---------------------------------------------------------------------------
typedef __attribute__((ext_vector_type(16))) __bf16 v16bf;
typedef __attribute__((ext_vector_type(8)))  __bf16 v8bf;
typedef __attribute__((ext_vector_type(4)))  __bf16 v4bf;
typedef __attribute__((ext_vector_type(8)))  float  v8f;

#define ACT_NONE 0
#define ACT_RELU 1
#define ACT_ELU 2
#define ACT_SIGMOID 3

#define GAT_ALPHA 0.2f
#define NEG_FILL -9.0e15f
#define INV_T 0.1f

// ---------------------------------------------------------------------------
// f32 -> bf16 conversion (grid stride)
// ---------------------------------------------------------------------------
__global__ void cvt_f32_bf16(const float* __restrict__ in, __bf16* __restrict__ out,
                             size_t n) {
  size_t i = (size_t)blockIdx.x * blockDim.x + threadIdx.x;
  size_t stride = (size_t)gridDim.x * blockDim.x;
  for (; i < n; i += stride) out[i] = (__bf16)in[i];
}

// ---------------------------------------------------------------------------
// Generic bf16 GEMM with WMMA: C[M,N] = act(A[M,K] @ B[K,N] + bias)
//   - A, B bf16 row-major; C f32; Cb optional bf16 mirror of C
//   - block tile 128x64, 8 waves, wave tile 32x32 (2x2 x v_wmma_f32_16x16x32_bf16)
//   - double-buffered LDS pipeline; vectorized staging on interior tiles
// ---------------------------------------------------------------------------
__global__ __launch_bounds__(256)
void gemm_bf16_wmma(const __bf16* __restrict__ A, const __bf16* __restrict__ B,
                    const float* __restrict__ bias,
                    float* __restrict__ C, __bf16* __restrict__ Cb,
                    int M, int K, int N, int act) {
  constexpr int BM = 128, BN = 64, BK = 32;
  __shared__ __bf16 sA[2][BM][BK];   // row-major [m][k]   : 2 x 8 KB
  __shared__ __bf16 sB[2][BN][BK];   // transposed [n][k]  : 2 x 4 KB

  const int tid  = threadIdx.x;
  const int lane = tid & 31;
  const int wave = tid >> 5;
  const int wm   = wave & 3;     // 4 waves along M
  const int wn   = wave >> 2;    // 2 waves along N
  const int half = lane >> 4;    // 0: lanes 0-15, 1: lanes 16-31
  const int l16  = lane & 15;

  const int mBase = blockIdx.y * BM;
  const int nBase = blockIdx.x * BN;

  const bool fullM = (mBase + BM <= M);
  const bool fullN = (nBase + BN <= N);
  const bool alnK  = ((K & 3) == 0);
  const bool alnN  = ((N & 3) == 0);

  // ---- stage one BK-tile into buffer `buf` (uniform fast/slow selection) ----
  auto stage = [&](int buf, int k0) {
    const bool fullK = (k0 + BK <= K);
    if (fullM && fullK && alnK) {
      // A: 128x32 elems as 8-byte chunks, 4 per thread, fully coalesced
#pragma unroll
      for (int i = 0; i < 4; ++i) {
        int idx4 = i * 256 + tid;
        int r = idx4 >> 3, c = (idx4 & 7) * 4;
        *(v4bf*)&sA[buf][r][c] =
            *(const v4bf*)&A[(size_t)(mBase + r) * K + (k0 + c)];
      }
    } else {
#pragma unroll
      for (int i = 0; i < 16; ++i) {
        int idx = i * 256 + tid;
        int r = idx >> 5, c = idx & 31;
        int gr = mBase + r, gk = k0 + c;
        __bf16 v = (__bf16)0.0f;
        if (gr < M && gk < K) v = A[(size_t)gr * K + gk];
        sA[buf][r][c] = v;
      }
    }
    if (fullN && fullK && alnN) {
      // B: 32x64 elems as 8-byte chunks along N, transposed into [n][k]
#pragma unroll
      for (int i = 0; i < 2; ++i) {
        int idx4 = i * 256 + tid;
        int kk = idx4 >> 4, nn = (idx4 & 15) * 4;
        v4bf v = *(const v4bf*)&B[(size_t)(k0 + kk) * N + (nBase + nn)];
        sB[buf][nn + 0][kk] = v[0];
        sB[buf][nn + 1][kk] = v[1];
        sB[buf][nn + 2][kk] = v[2];
        sB[buf][nn + 3][kk] = v[3];
      }
    } else {
#pragma unroll
      for (int i = 0; i < 8; ++i) {
        int idx = i * 256 + tid;
        int kk = idx >> 6, nn = idx & 63;
        int gk = k0 + kk, gn = nBase + nn;
        __bf16 v = (__bf16)0.0f;
        if (gk < K && gn < N) v = B[(size_t)gk * N + gn];
        sB[buf][nn][kk] = v;
      }
    }
  };

  v8f c[2][2] = {};
  const int nTiles = (K + BK - 1) / BK;
  int buf = 0;

  stage(0, 0);

  for (int t = 0; t < nTiles; ++t) {
    __syncthreads();

    // ---- fragments from current buffer (ds_load_b128 pairs)
    // A lane: row M=l16; lanes 0-15 hold K{0..7,16..23}, lanes 16-31 K{8..15,24..31}
    v16bf afrag[2], bfrag[2];
#pragma unroll
    for (int mi = 0; mi < 2; ++mi) {
      int row = wm * 32 + mi * 16 + l16;
      const v8bf* p0 = (const v8bf*)&sA[buf][row][half * 8];
      const v8bf* p1 = (const v8bf*)&sA[buf][row][16 + half * 8];
      afrag[mi] = __builtin_shufflevector(*p0, *p1, 0, 1, 2, 3, 4, 5, 6, 7,
                                          8, 9, 10, 11, 12, 13, 14, 15);
    }
    // B lane: col N=l16; lanes 0-15 hold K{0..7,16..23}, lanes 16-31 K{8..15,24..31}
#pragma unroll
    for (int nj = 0; nj < 2; ++nj) {
      int col = wn * 32 + nj * 16 + l16;
      const v8bf* q0 = (const v8bf*)&sB[buf][col][half * 8];
      const v8bf* q1 = (const v8bf*)&sB[buf][col][16 + half * 8];
      bfrag[nj] = __builtin_shufflevector(*q0, *q1, 0, 1, 2, 3, 4, 5, 6, 7,
                                          8, 9, 10, 11, 12, 13, 14, 15);
    }

    // ---- overlap: stage next tile into the other buffer
    if (t + 1 < nTiles) stage(buf ^ 1, (t + 1) * BK);

    // prefetch two tiles ahead (lowers to global_prefetch_b8)
    if ((t + 2) * BK < K && fullM) {
      int gr = mBase + (tid >> 1);
      __builtin_prefetch(&A[(size_t)gr * K + (t + 2) * BK], 0, 1);
    }

#pragma unroll
    for (int mi = 0; mi < 2; ++mi)
#pragma unroll
      for (int nj = 0; nj < 2; ++nj)
        c[mi][nj] = __builtin_amdgcn_wmma_f32_16x16x32_bf16(
            false, afrag[mi], false, bfrag[nj], (short)0, c[mi][nj], false, false);

    buf ^= 1;
  }

  // ---- epilogue: C layout: VGPR r -> M = r + half*8 ; N = l16
  if (fullM && fullN) {
#pragma unroll
    for (int mi = 0; mi < 2; ++mi) {
#pragma unroll
      for (int nj = 0; nj < 2; ++nj) {
        int col = nBase + wn * 32 + nj * 16 + l16;
        float bv = bias ? bias[col] : 0.f;
#pragma unroll
        for (int r = 0; r < 8; ++r) {
          int row = mBase + wm * 32 + mi * 16 + half * 8 + r;
          float v = c[mi][nj][r] + bv;
          if (act == ACT_RELU)         v = v > 0.f ? v : 0.f;
          else if (act == ACT_ELU)     v = v > 0.f ? v : (__expf(v) - 1.f);
          else if (act == ACT_SIGMOID) v = 1.f / (1.f + __expf(-v));
          C[(size_t)row * N + col] = v;
          if (Cb) Cb[(size_t)row * N + col] = (__bf16)v;
        }
      }
    }
  } else {
#pragma unroll
    for (int mi = 0; mi < 2; ++mi) {
#pragma unroll
      for (int nj = 0; nj < 2; ++nj) {
        int col = nBase + wn * 32 + nj * 16 + l16;
#pragma unroll
        for (int r = 0; r < 8; ++r) {
          int row = mBase + wm * 32 + mi * 16 + half * 8 + r;
          if (row < M && col < N) {
            float v = c[mi][nj][r];
            if (bias) v += bias[col];
            if (act == ACT_RELU)         v = v > 0.f ? v : 0.f;
            else if (act == ACT_ELU)     v = v > 0.f ? v : (__expf(v) - 1.f);
            else if (act == ACT_SIGMOID) v = 1.f / (1.f + __expf(-v));
            C[(size_t)row * N + col] = v;
            if (Cb) Cb[(size_t)row * N + col] = (__bf16)v;
          }
        }
      }
    }
  }
}

// ---------------------------------------------------------------------------
// f1 = Wh @ a[:d], f2 = Wh @ a[d:]   (thread per row)
// ---------------------------------------------------------------------------
__global__ void gat_rowdots(const float* __restrict__ Wh, const float* __restrict__ a,
                            float* __restrict__ f1, float* __restrict__ f2,
                            int M, int d) {
  int row = blockIdx.x * blockDim.x + threadIdx.x;
  if (row >= M) return;
  const float* w = Wh + (size_t)row * d;
  float s1 = 0.f, s2 = 0.f;
  for (int j = 0; j < d; ++j) {
    float wv = w[j];
    s1 = fmaf(wv, a[j], s1);
    s2 = fmaf(wv, a[d + j], s2);
  }
  f1[row] = s1;
  f2[row] = s2;
}

// ---------------------------------------------------------------------------
// One block per row: att[i,:] = softmax(mask(leakyrelu(f1[i]+f2[:]), adj[i,:]))
// N = 6144 fixed, 256 threads, 24 cols/thread kept in registers -> bf16 out
// ---------------------------------------------------------------------------
__global__ __launch_bounds__(256)
void gat_attention(const float* __restrict__ f1, const float* __restrict__ f2,
                   const float* __restrict__ adj, __bf16* __restrict__ att) {
  constexpr int NN = 6144, CPT = 24;
  const int row = blockIdx.x;
  const int tid = threadIdx.x;
  __shared__ float red[256];

  float e[CPT];
  const float fi = f1[row];
  const float* arow = adj + (size_t)row * NN;

  float mx = -3.4e38f;
#pragma unroll
  for (int i = 0; i < CPT; ++i) {
    int col = tid + i * 256;
    float v = fi + f2[col];
    v = v > 0.f ? v : GAT_ALPHA * v;         // leaky-relu
    v = arow[col] > 0.f ? v : NEG_FILL;      // adjacency mask
    e[i] = v;
    mx = fmaxf(mx, v);
  }
  red[tid] = mx; __syncthreads();
  for (int s = 128; s > 0; s >>= 1) {
    if (tid < s) red[tid] = fmaxf(red[tid], red[tid + s]);
    __syncthreads();
  }
  mx = red[0]; __syncthreads();

  float sum = 0.f;
#pragma unroll
  for (int i = 0; i < CPT; ++i) {
    float p = __expf(e[i] - mx);
    e[i] = p;
    sum += p;
  }
  red[tid] = sum; __syncthreads();
  for (int s = 128; s > 0; s >>= 1) {
    if (tid < s) red[tid] += red[tid + s];
    __syncthreads();
  }
  float inv = 1.f / red[0];
#pragma unroll
  for (int i = 0; i < CPT; ++i)
    att[(size_t)row * NN + tid + i * 256] = (__bf16)(e[i] * inv);
}

// ---------------------------------------------------------------------------
// y[M, 2d] = concat(h[M,d], k[M,d]) as bf16
// ---------------------------------------------------------------------------
__global__ void concat_bf16(const float* __restrict__ h, const float* __restrict__ k,
                            __bf16* __restrict__ y, int M, int d) {
  size_t n = (size_t)M * 2 * d;
  size_t i = (size_t)blockIdx.x * blockDim.x + threadIdx.x;
  size_t stride = (size_t)gridDim.x * blockDim.x;
  for (; i < n; i += stride) {
    size_t row = i / (2 * d);
    int c = (int)(i - row * (2 * d));
    float v = (c < d) ? h[row * d + c] : k[row * d + (c - d)];
    y[i] = (__bf16)v;
  }
}

// ---------------------------------------------------------------------------
// fusion weights: wsum += sum_rows softmax(a3/T)     (a3 is [M,2])
// ---------------------------------------------------------------------------
__global__ void zero2(float* w) {
  if (threadIdx.x < 2) w[threadIdx.x] = 0.f;
}

__global__ __launch_bounds__(256)
void fuse_wsum(const float* __restrict__ a3, float* __restrict__ wsum, int M) {
  int row = blockIdx.x * 256 + threadIdx.x;
  float p0 = 0.f, p1 = 0.f;
  if (row < M) {
    float x0 = a3[2 * row] * INV_T, x1 = a3[2 * row + 1] * INV_T;
    float m = fmaxf(x0, x1);
    float e0 = __expf(x0 - m), e1 = __expf(x1 - m);
    float s = e0 + e1;
    p0 = e0 / s; p1 = e1 / s;
  }
  __shared__ float r0[256], r1[256];
  r0[threadIdx.x] = p0; r1[threadIdx.x] = p1; __syncthreads();
  for (int s = 128; s > 0; s >>= 1) {
    if (threadIdx.x < s) {
      r0[threadIdx.x] += r0[threadIdx.x + s];
      r1[threadIdx.x] += r1[threadIdx.x + s];
    }
    __syncthreads();
  }
  if (threadIdx.x == 0) {
    atomicAdd(&wsum[0], r0[0]);
    atomicAdd(&wsum[1], r1[0]);
  }
}

// h_out = w0*h + w1*k  with w = wsum/M ; optional bf16 mirror
__global__ void fuse_combine(const float* __restrict__ h, const float* __restrict__ k,
                             const float* __restrict__ wsum, float* __restrict__ hout,
                             __bf16* __restrict__ hbout, size_t n, float invM) {
  float w0 = wsum[0] * invM, w1 = wsum[1] * invM;
  size_t i = (size_t)blockIdx.x * blockDim.x + threadIdx.x;
  size_t stride = (size_t)gridDim.x * blockDim.x;
  for (; i < n; i += stride) {
    float v = w0 * h[i] + w1 * k[i];
    hout[i] = v;
    if (hbout) hbout[i] = (__bf16)v;
  }
}

// ---------------------------------------------------------------------------
// predict = softmax(h[M,32] @ lin_w[32,10] + lin_b, axis=1)  (thread per row)
// ---------------------------------------------------------------------------
__global__ void head_predict(const float* __restrict__ h, const float* __restrict__ lw,
                             const float* __restrict__ lb, float* __restrict__ out,
                             int M) {
  int row = blockIdx.x * blockDim.x + threadIdx.x;
  if (row >= M) return;
  float hv[32];
#pragma unroll
  for (int j = 0; j < 32; ++j) hv[j] = h[(size_t)row * 32 + j];
  float o[10];
  float m = -3.4e38f;
#pragma unroll
  for (int kk = 0; kk < 10; ++kk) {
    float s = lb[kk];
#pragma unroll
    for (int j = 0; j < 32; ++j) s = fmaf(hv[j], lw[j * 10 + kk], s);
    o[kk] = s;
    m = fmaxf(m, s);
  }
  float sum = 0.f;
#pragma unroll
  for (int kk = 0; kk < 10; ++kk) { o[kk] = __expf(o[kk] - m); sum += o[kk]; }
  float inv = 1.f / sum;
#pragma unroll
  for (int kk = 0; kk < 10; ++kk) out[(size_t)row * 10 + kk] = o[kk] * inv;
}

// ---------------------------------------------------------------------------
// student-t q: q[i,k] = 1/(1+||z_i-mu_k||^2), normalized (V=1)
// ---------------------------------------------------------------------------
__global__ void head_q(const float* __restrict__ z, const float* __restrict__ cl,
                       float* __restrict__ out, int M) {
  int row = blockIdx.x * blockDim.x + threadIdx.x;
  if (row >= M) return;
  float zv[32];
#pragma unroll
  for (int j = 0; j < 32; ++j) zv[j] = z[(size_t)row * 32 + j];
  float qv[10];
  float sum = 0.f;
#pragma unroll
  for (int kk = 0; kk < 10; ++kk) {
    float d2 = 0.f;
#pragma unroll
    for (int j = 0; j < 32; ++j) {
      float t = zv[j] - cl[kk * 32 + j];
      d2 = fmaf(t, t, d2);
    }
    qv[kk] = 1.f / (1.f + d2);
    sum += qv[kk];
  }
  float inv = 1.f / sum;
#pragma unroll
  for (int kk = 0; kk < 10; ++kk) out[(size_t)row * 10 + kk] = qv[kk] * inv;
}

// ---------------------------------------------------------------------------
// Orchestration
// ---------------------------------------------------------------------------
extern "C" void kernel_launch(void* const* d_in, const int* in_sizes, int n_in,
                              void* d_out, int out_size, void* d_ws, size_t ws_size,
                              hipStream_t stream) {
  (void)in_sizes; (void)n_in; (void)out_size; (void)ws_size;
  const int M = 6144, NIN = 2000, D1 = 512, D2 = 256, DZ = 32;

  // ---- inputs (setup_inputs dict order) ----
  const float* x        = (const float*)d_in[0];
  const float* adj      = (const float*)d_in[1];
  const float* enc1_w   = (const float*)d_in[2];  const float* enc1_b = (const float*)d_in[3];
  const float* enc2_w   = (const float*)d_in[4];  const float* enc2_b = (const float*)d_in[5];
  const float* zl_w     = (const float*)d_in[6];  const float* zl_b   = (const float*)d_in[7];
  const float* dec1_w   = (const float*)d_in[8];  const float* dec1_b = (const float*)d_in[9];
  const float* dec2_w   = (const float*)d_in[10]; const float* dec2_b = (const float*)d_in[11];
  const float* xbar_w   = (const float*)d_in[12]; const float* xbar_b = (const float*)d_in[13];
  const float* gat1_W   = (const float*)d_in[14]; const float* gat1_a = (const float*)d_in[15];
  const float* gat2_W   = (const float*)d_in[16]; const float* gat2_a = (const float*)d_in[17];
  const float* gat3_W   = (const float*)d_in[18]; const float* gat3_a = (const float*)d_in[19];
  const float* f1c1_w   = (const float*)d_in[20]; const float* f1c1_b = (const float*)d_in[21];
  const float* f1c2_w   = (const float*)d_in[22]; const float* f1c2_b = (const float*)d_in[23];
  const float* f1c3_w   = (const float*)d_in[24]; const float* f1c3_b = (const float*)d_in[25];
  const float* f2c1_w   = (const float*)d_in[26]; const float* f2c1_b = (const float*)d_in[27];
  const float* f2c2_w   = (const float*)d_in[28]; const float* f2c2_b = (const float*)d_in[29];
  const float* f2c3_w   = (const float*)d_in[30]; const float* f2c3_b = (const float*)d_in[31];
  const float* f3c1_w   = (const float*)d_in[32]; const float* f3c1_b = (const float*)d_in[33];
  const float* f3c2_w   = (const float*)d_in[34]; const float* f3c2_b = (const float*)d_in[35];
  const float* f3c3_w   = (const float*)d_in[36]; const float* f3c3_b = (const float*)d_in[37];
  const float* lin_w    = (const float*)d_in[38]; const float* lin_b  = (const float*)d_in[39];
  const float* cluster  = (const float*)d_in[40];

  // ---- outputs: x_bar | q | predict | z (flat, in return order) ----
  float* out_xbar = (float*)d_out;
  float* out_q    = out_xbar + (size_t)M * NIN;
  float* out_pred = out_q + (size_t)M * 10;
  float* out_z    = out_pred + (size_t)M * 10;

  // ---- workspace carve-out ----
  char* ws = (char*)d_ws;
  size_t off = 0;
  auto alloc = [&](size_t bytes) -> char* {
    char* p = ws + off;
    off += (bytes + 255) & ~(size_t)255;
    return p;
  };

  __bf16* xb      = (__bf16*)alloc((size_t)M * NIN * 2);
  __bf16* attb    = (__bf16*)alloc((size_t)M * M * 2);

  __bf16* enc1_wb = (__bf16*)alloc((size_t)NIN * D1 * 2);
  __bf16* enc2_wb = (__bf16*)alloc((size_t)D1 * D2 * 2);
  __bf16* zl_wb   = (__bf16*)alloc((size_t)D2 * DZ * 2);
  __bf16* dec1_wb = (__bf16*)alloc((size_t)DZ * D2 * 2);
  __bf16* dec2_wb = (__bf16*)alloc((size_t)D2 * D1 * 2);
  __bf16* xbar_wb = (__bf16*)alloc((size_t)D1 * NIN * 2);
  __bf16* gat1_Wb = (__bf16*)alloc((size_t)NIN * D1 * 2);
  __bf16* gat2_Wb = (__bf16*)alloc((size_t)D1 * D2 * 2);
  __bf16* gat3_Wb = (__bf16*)alloc((size_t)D2 * DZ * 2);
  __bf16* f1c1_wb = (__bf16*)alloc((size_t)1024 * 500 * 2);
  __bf16* f1c2_wb = (__bf16*)alloc((size_t)500 * 100 * 2);
  __bf16* f1c3_wb = (__bf16*)alloc((size_t)100 * 2 * 2);
  __bf16* f2c1_wb = (__bf16*)alloc((size_t)512 * 500 * 2);
  __bf16* f2c2_wb = (__bf16*)alloc((size_t)500 * 100 * 2);
  __bf16* f2c3_wb = (__bf16*)alloc((size_t)100 * 2 * 2);
  __bf16* f3c1_wb = (__bf16*)alloc((size_t)64 * 500 * 2);
  __bf16* f3c2_wb = (__bf16*)alloc((size_t)500 * 100 * 2);
  __bf16* f3c3_wb = (__bf16*)alloc((size_t)100 * 2 * 2);

  float*  t1  = (float*)alloc((size_t)M * D1 * 4);
  __bf16* t1b = (__bf16*)alloc((size_t)M * D1 * 2);
  float*  t2  = (float*)alloc((size_t)M * D2 * 4);
  __bf16* t2b = (__bf16*)alloc((size_t)M * D2 * 2);
  __bf16* zb  = (__bf16*)alloc((size_t)M * DZ * 2);
  float*  dd1 = (float*)alloc((size_t)M * D2 * 4);
  __bf16* dd1b= (__bf16*)alloc((size_t)M * D2 * 2);
  float*  dd2 = (float*)alloc((size_t)M * D1 * 4);
  __bf16* dd2b= (__bf16*)alloc((size_t)M * D1 * 2);

  float*  Wh  = (float*)alloc((size_t)M * D1 * 4);
  __bf16* Whb = (__bf16*)alloc((size_t)M * D1 * 2);
  float*  f1v = (float*)alloc((size_t)M * 4);
  float*  f2v = (float*)alloc((size_t)M * 4);

  float*  h   = (float*)alloc((size_t)M * D1 * 4);   // GAT output
  float*  h2  = (float*)alloc((size_t)M * D1 * 4);   // fuse output
  __bf16* hb  = (__bf16*)alloc((size_t)M * D1 * 2);  // bf16 of fuse output
  __bf16* y   = (__bf16*)alloc((size_t)M * 1024 * 2);

  float*  a1  = (float*)alloc((size_t)M * 500 * 4);
  __bf16* a1b = (__bf16*)alloc((size_t)M * 500 * 2);
  float*  a2  = (float*)alloc((size_t)M * 100 * 4);
  __bf16* a2b = (__bf16*)alloc((size_t)M * 100 * 2);
  float*  a3  = (float*)alloc((size_t)M * 2 * 4);
  float*  wsum= (float*)alloc(2 * 4);

  // ---- launch helpers ----
  auto cvt = [&](const float* in, __bf16* out, size_t n) {
    size_t b = (n + 255) / 256;
    if (b > 4096) b = 4096;
    cvt_f32_bf16<<<dim3((unsigned)b), dim3(256), 0, stream>>>(in, out, n);
  };
  auto gemm = [&](const __bf16* A, const __bf16* B, const float* bias,
                  float* C, __bf16* Cb, int Mm, int Kk, int Nn, int act) {
    dim3 g((Nn + 63) / 64, (Mm + 127) / 128);
    gemm_bf16_wmma<<<g, dim3(256), 0, stream>>>(A, B, bias, C, Cb, Mm, Kk, Nn, act);
  };
  auto grid1d = [](int n) { return dim3((n + 255) / 256); };

  // ---- convert input + weights to bf16 ----
  cvt(x, xb, (size_t)M * NIN);
  cvt(enc1_w, enc1_wb, (size_t)NIN * D1);
  cvt(enc2_w, enc2_wb, (size_t)D1 * D2);
  cvt(zl_w, zl_wb, (size_t)D2 * DZ);
  cvt(dec1_w, dec1_wb, (size_t)DZ * D2);
  cvt(dec2_w, dec2_wb, (size_t)D2 * D1);
  cvt(xbar_w, xbar_wb, (size_t)D1 * NIN);
  cvt(gat1_W, gat1_Wb, (size_t)NIN * D1);
  cvt(gat2_W, gat2_Wb, (size_t)D1 * D2);
  cvt(gat3_W, gat3_Wb, (size_t)D2 * DZ);
  cvt(f1c1_w, f1c1_wb, (size_t)1024 * 500);
  cvt(f1c2_w, f1c2_wb, (size_t)500 * 100);
  cvt(f1c3_w, f1c3_wb, (size_t)100 * 2);
  cvt(f2c1_w, f2c1_wb, (size_t)512 * 500);
  cvt(f2c2_w, f2c2_wb, (size_t)500 * 100);
  cvt(f2c3_w, f2c3_wb, (size_t)100 * 2);
  cvt(f3c1_w, f3c1_wb, (size_t)64 * 500);
  cvt(f3c2_w, f3c2_wb, (size_t)500 * 100);
  cvt(f3c3_w, f3c3_wb, (size_t)100 * 2);

  // ---- autoencoder ----
  gemm(xb,   enc1_wb, enc1_b, t1,      t1b,  M, NIN, D1, ACT_RELU);
  gemm(t1b,  enc2_wb, enc2_b, t2,      t2b,  M, D1,  D2, ACT_RELU);
  gemm(t2b,  zl_wb,   zl_b,   out_z,   zb,   M, D2,  DZ, ACT_NONE);
  gemm(zb,   dec1_wb, dec1_b, dd1,     dd1b, M, DZ,  D2, ACT_RELU);
  gemm(dd1b, dec2_wb, dec2_b, dd2,     dd2b, M, D2,  D1, ACT_RELU);
  gemm(dd2b, xbar_wb, xbar_b, out_xbar, nullptr, M, D1, NIN, ACT_NONE);

  // ---- GAT layer 1 (in: x [M,2000], d=512) + fuse with t1 ----
  gemm(xb, gat1_Wb, nullptr, Wh, Whb, M, NIN, D1, ACT_NONE);
  gat_rowdots<<<grid1d(M), 256, 0, stream>>>(Wh, gat1_a, f1v, f2v, M, D1);
  gat_attention<<<dim3(M), 256, 0, stream>>>(f1v, f2v, adj, attb);
  gemm(attb, Whb, nullptr, h, nullptr, M, M, D1, ACT_ELU);
  concat_bf16<<<4096, 256, 0, stream>>>(h, t1, y, M, D1);
  gemm(y,   f1c1_wb, f1c1_b, a1, a1b, M, 1024, 500, ACT_RELU);
  gemm(a1b, f1c2_wb, f1c2_b, a2, a2b, M, 500,  100, ACT_RELU);
  gemm(a2b, f1c3_wb, f1c3_b, a3, nullptr, M, 100, 2, ACT_SIGMOID);
  zero2<<<1, 32, 0, stream>>>(wsum);
  fuse_wsum<<<grid1d(M), 256, 0, stream>>>(a3, wsum, M);
  fuse_combine<<<4096, 256, 0, stream>>>(h, t1, wsum, h2, hb, (size_t)M * D1,
                                         1.0f / (float)M);

  // ---- GAT layer 2 (in: h [M,512], d=256) + fuse with t2 ----
  gemm(hb, gat2_Wb, nullptr, Wh, Whb, M, D1, D2, ACT_NONE);
  gat_rowdots<<<grid1d(M), 256, 0, stream>>>(Wh, gat2_a, f1v, f2v, M, D2);
  gat_attention<<<dim3(M), 256, 0, stream>>>(f1v, f2v, adj, attb);
  gemm(attb, Whb, nullptr, h, nullptr, M, M, D2, ACT_ELU);
  concat_bf16<<<4096, 256, 0, stream>>>(h, t2, y, M, D2);
  gemm(y,   f2c1_wb, f2c1_b, a1, a1b, M, 512, 500, ACT_RELU);
  gemm(a1b, f2c2_wb, f2c2_b, a2, a2b, M, 500, 100, ACT_RELU);
  gemm(a2b, f2c3_wb, f2c3_b, a3, nullptr, M, 100, 2, ACT_SIGMOID);
  zero2<<<1, 32, 0, stream>>>(wsum);
  fuse_wsum<<<grid1d(M), 256, 0, stream>>>(a3, wsum, M);
  fuse_combine<<<4096, 256, 0, stream>>>(h, t2, wsum, h2, hb, (size_t)M * D2,
                                         1.0f / (float)M);

  // ---- GAT layer 3 (in: h [M,256], d=32) + fuse with z ----
  gemm(hb, gat3_Wb, nullptr, Wh, Whb, M, D2, DZ, ACT_NONE);
  gat_rowdots<<<grid1d(M), 256, 0, stream>>>(Wh, gat3_a, f1v, f2v, M, DZ);
  gat_attention<<<dim3(M), 256, 0, stream>>>(f1v, f2v, adj, attb);
  gemm(attb, Whb, nullptr, h, nullptr, M, M, DZ, ACT_ELU);
  concat_bf16<<<1024, 256, 0, stream>>>(h, out_z, y, M, DZ);
  gemm(y,   f3c1_wb, f3c1_b, a1, a1b, M, 64,  500, ACT_RELU);
  gemm(a1b, f3c2_wb, f3c2_b, a2, a2b, M, 500, 100, ACT_RELU);
  gemm(a2b, f3c3_wb, f3c3_b, a3, nullptr, M, 100, 2, ACT_SIGMOID);
  zero2<<<1, 32, 0, stream>>>(wsum);
  fuse_wsum<<<grid1d(M), 256, 0, stream>>>(a3, wsum, M);
  fuse_combine<<<1024, 256, 0, stream>>>(h, out_z, wsum, h2, nullptr,
                                         (size_t)M * DZ, 1.0f / (float)M);

  // ---- heads ----
  head_predict<<<grid1d(M), 256, 0, stream>>>(h2, lin_w, lin_b, out_pred, M);
  head_q<<<grid1d(M), 256, 0, stream>>>(out_z, cluster, out_q, M);
}
